// SelfAttn2d_57080115364820
// MI455X (gfx1250) — matrix operand order repaired and checked by reading, not compiled, not run on hardware
//
#include <hip/hip_runtime.h>

// ---------- types ----------
typedef __attribute__((ext_vector_type(16))) __bf16 v16bf;
typedef __attribute__((ext_vector_type(2)))  __bf16 v2bf;
typedef __attribute__((ext_vector_type(8)))  float  v8f;
typedef __attribute__((ext_vector_type(4)))  unsigned v4u;
typedef __attribute__((ext_vector_type(8)))  int      v8i;
typedef __attribute__((ext_vector_type(4)))  int      v4i;

union AFrag { v16bf v; unsigned u[8]; unsigned short us[16]; };
union BFrag { v16bf v; unsigned u[8]; uint4 q[2]; unsigned short us[16]; };

// native fptrunc -> v_cvt_pk_bf16_f32 on gfx1250 (verified round 3)
static __device__ __forceinline__ unsigned short f2bf(float f) {
    return __builtin_bit_cast(unsigned short, (__bf16)f);
}
static __device__ __forceinline__ unsigned pack_bf16(float lo, float hi) {
    v2bf p = { (__bf16)lo, (__bf16)hi };
    return __builtin_bit_cast(unsigned, p);
}

static __device__ __forceinline__ v8f wmma_bf16(v16bf a, v16bf b, v8f c) {
    return __builtin_amdgcn_wmma_f32_16x16x32_bf16(
        /*neg_a=*/false, a, /*neg_b=*/false, b,
        /*c_mod=*/(short)0, c, /*reuse_a=*/false, /*reuse_b=*/false);
}

#define NTOK 4096
#define CCH  256
#define HAVE_TDM __has_builtin(__builtin_amdgcn_tensor_load_to_lds)

// ---------------------------------------------------------------------------
// Kernel 0: convert projection weights f32 -> bf16 once (row-major preserved).
// ---------------------------------------------------------------------------
__global__ void __launch_bounds__(256)
wconv_kernel(const float* __restrict__ Wq, const float* __restrict__ Wk,
             const float* __restrict__ Wv,
             unsigned* __restrict__ wqb, unsigned* __restrict__ wkb,
             unsigned* __restrict__ wvb)
{
    const int i = blockIdx.x * blockDim.x + threadIdx.x;   // pair index
    if (i < 4096)  wqb[i] = pack_bf16(Wq[2 * i], Wq[2 * i + 1]);
    if (i < 4096)  wkb[i] = pack_bf16(Wk[2 * i], Wk[2 * i + 1]);
    if (i < 32768) wvb[i] = pack_bf16(Wv[2 * i], Wv[2 * i + 1]);
}

// ---------------------------------------------------------------------------
// Kernel 1: projections.  q/k -> bf16 [b][n][32], v -> bf16 transposed [b][c][n]
// GEMM: out[n,d] = sum_c x[b,c,n] * W[d,c]  via A = X^T (16x32), B = W^T (32x16)
// ---------------------------------------------------------------------------
__global__ void __launch_bounds__(128)
proj_kernel(const float* __restrict__ x,
            const unsigned short* __restrict__ wqb, const float* __restrict__ bq,
            const unsigned short* __restrict__ wkb, const float* __restrict__ bk,
            const unsigned short* __restrict__ wvb, const float* __restrict__ bv,
            unsigned short* __restrict__ qws,
            unsigned short* __restrict__ kws,
            unsigned short* __restrict__ vws)
{
    const int wave = threadIdx.x >> 5;
    const int lane = threadIdx.x & 31;
    const int col  = lane & 15;
    const int hi   = (lane >> 4) & 1;

    const int tile  = blockIdx.x;
    const int b     = tile >> 6;               // 64 tiles per batch
    const int nbase = (tile & 63) * 64 + wave * 16;
    const int n     = nbase + col;             // this lane's pixel row (A rows)
    const int kbaseA = hi * 8;
    const int kb     = hi * 16;

    const float* xb = x + (size_t)b * CCH * NTOK;

    v8f accq[2], acck[2], accv[16];
#pragma unroll
    for (int t = 0; t < 2; ++t)
#pragma unroll
        for (int i = 0; i < 8; ++i) { accq[t][i] = 0.f; acck[t][i] = 0.f; }
#pragma unroll
    for (int t = 0; t < 16; ++t)
#pragma unroll
        for (int i = 0; i < 8; ++i) accv[t][i] = 0.f;

    for (int cc = 0; cc < 8; ++cc) {
        const int c0 = cc * 32;
        // ---- A fragment: X^T tile, 16 pixels x 32 channels (bf16) ----
        AFrag a;
#pragma unroll
        for (int j = 0; j < 8; ++j) {
            const int K0 = kb + 2 * j;
            const float f0 = xb[(size_t)(c0 + K0)     * NTOK + n];
            const float f1 = xb[(size_t)(c0 + K0 + 1) * NTOK + n];
            a.u[j] = pack_bf16(f0, f1);
        }
        // ---- B fragments: bf16 weights, 16 contiguous elems along c ----
        auto load_b = [&](const unsigned short* wb, int d0) -> BFrag {
            BFrag bf;
            const uint4* p =
                (const uint4*)(wb + (size_t)(d0 + col) * CCH + c0 + kb);
            bf.q[0] = p[0]; bf.q[1] = p[1];
            return bf;
        };
        // software pipeline: load tile t+1 before WMMA on tile t
        BFrag cur = load_b(wqb, 0);
#define PSTEP(ACC, NW, ND) { BFrag nxt = load_b(NW, ND); \
        ACC = wmma_bf16(a.v, cur.v, ACC); cur = nxt; }
        PSTEP(accq[0], wqb, 16)
        PSTEP(accq[1], wkb, 0)
        PSTEP(acck[0], wkb, 16)
        PSTEP(acck[1], wvb, 0)
#pragma unroll
        for (int t = 0; t < 15; ++t) PSTEP(accv[t], wvb, (t + 1) * 16)
        accv[15] = wmma_bf16(a.v, cur.v, accv[15]);
#undef PSTEP
    }

    // ---- store: D layout row = r + hi*8, col = lane&15 ----
#pragma unroll
    for (int r = 0; r < 8; ++r) {
        const int np = nbase + r + hi * 8;
#pragma unroll
        for (int t = 0; t < 2; ++t) {
            const int d = t * 16 + col;
            qws[((size_t)b * NTOK + np) * 32 + d] = f2bf(accq[t][r] + bq[d]);
            kws[((size_t)b * NTOK + np) * 32 + d] = f2bf(acck[t][r] + bk[d]);
        }
#pragma unroll
        for (int t = 0; t < 16; ++t) {
            const int c = t * 16 + col;
            vws[((size_t)b * CCH + c) * NTOK + np] = f2bf(accv[t][r] + bv[c]);
        }
    }
}

// ---------------------------------------------------------------------------
// TDM: stage one V tile (32 kv x 256 c, bf16, compacted rows of 32 elems)
// into LDS.  D# per CDNA5 ISA ch.8: group0 = {count, lds_addr, gaddr, type},
// group1 = {data_size=2B, tensor_dim0=4096, tensor_dim1=256, tile 32x256,
// dim0_stride=4096}.  2D tensor -> remaining groups zero.
// clang-23 toolchain: 6-arg builtin (g0 u32x4, g1 i32x8, i32x4, i32x4, i32x8, cpol)
// ---------------------------------------------------------------------------
#if HAVE_TDM
static __device__ __forceinline__ void tdm_load_vtile(const unsigned short* gsrc,
                                                      unsigned lds_off) {
    const unsigned long long ga = (unsigned long long)(size_t)gsrc;
    v4u g0 = { 1u,                                  // count=1, user mode
               lds_off,                             // lds_addr (bytes)
               (unsigned)ga,                        // global_addr[31:0]
               (unsigned)(ga >> 32) | 0x80000000u };// global_addr[56:32] | type=2
    v8i g1 = { (int)(1u << 16),          // data_size = 1 -> 2 bytes
               (int)(4096u << 16),       // tensor_dim0[15:0] = 4096 (<<16)
               (int)(256u  << 16),       // tensor_dim0[31:16]=0 | tensor_dim1=256
               (int)(32u   << 16),       // tensor_dim1[31:16]=0 | tile_dim0=32
               (int)256,                 // tile_dim1=256 | tile_dim2=0
               (int)4096,                // tensor_dim0_stride[31:0]
               0, 0 };                   // stride hi, dim1_stride (unused 2D)
    v4i gz4 = { 0, 0, 0, 0 };
    v8i gz8 = { 0, 0, 0, 0, 0, 0, 0, 0 };
    __builtin_amdgcn_tensor_load_to_lds(g0, g1, gz4, gz4, gz8, 0);
}
#endif

// ---------------------------------------------------------------------------
// Kernel 2: flash attention.  16 query rows per wave, kv blocks of 32.
// V tiles staged block-wide by the TDM (double buffered), K pipelined per wave.
// ---------------------------------------------------------------------------
__global__ void __launch_bounds__(128)
attn_kernel(const float* __restrict__ x, const float* __restrict__ gamma,
            const unsigned short* __restrict__ qws,
            const unsigned short* __restrict__ kws,
            const unsigned short* __restrict__ vws,
            float* __restrict__ out)
{
    __shared__ __align__(16) unsigned short plds[4][16][32];
#if HAVE_TDM
    __shared__ __align__(16) unsigned short vlds[2][CCH][32];
#endif

    const int wave = threadIdx.x >> 5;
    const int lane = threadIdx.x & 31;
    const int col  = lane & 15;
    const int hi   = (lane >> 4) & 1;

    const int b     = blockIdx.x >> 6;
    const int qbase = (blockIdx.x & 63) * 64 + wave * 16;
    const int kbaseA = hi * 8;
    const int kb     = hi * 16;

    // ---- load Q as A fragment (16 rows x 32 d), rows contiguous in ws ----
    AFrag qa;
    {
        const unsigned* qrow =
            (const unsigned*)(qws + ((size_t)b * NTOK + qbase + col) * 32);
#pragma unroll
        for (int j = 0; j < 8; ++j) {
            const int K0 = kb + 2 * j;
            qa.u[j] = qrow[K0 >> 1];
        }
    }

    v8f o[16];
#pragma unroll
    for (int t = 0; t < 16; ++t)
#pragma unroll
        for (int i = 0; i < 8; ++i) o[t][i] = 0.f;

    float m[8], l[8];
#pragma unroll
    for (int r = 0; r < 8; ++r) { m[r] = -1e30f; l[r] = 0.f; }

    v8f zf;
#pragma unroll
    for (int i = 0; i < 8; ++i) zf[i] = 0.f;

    const unsigned short* kbase_p = kws + (size_t)b * NTOK * 32;
    const unsigned short* vbase_p = vws + (size_t)b * CCH * NTOK;

    auto load_k = [&](int kv, BFrag& f) {
        const uint4* p = (const uint4*)(kbase_p + (size_t)(kv + col) * 32 + kb);
        f.q[0] = p[0]; f.q[1] = p[1];
    };

    BFrag kf0, kf1;
    load_k(0, kf0);
    load_k(16, kf1);

    const int NB = NTOK / 32;
#if HAVE_TDM
    if (wave == 0)
        tdm_load_vtile(vbase_p, (unsigned)(size_t)&vlds[0][0][0]);
#endif

    for (int jb = 0; jb < NB; ++jb) {
        const int kv0 = jb * 32;
        const int buf = jb & 1;

#if HAVE_TDM
        if (wave == 0) {
            __builtin_amdgcn_s_wait_tensorcnt(0);          // tile jb landed
            if (jb + 1 < NB)                               // prefetch jb+1
                tdm_load_vtile(vbase_p + (size_t)(kv0 + 32),
                               (unsigned)(size_t)&vlds[buf ^ 1][0][0]);
        }
        __syncthreads();                                   // tile jb visible to all
#endif

        v8f s0 = wmma_bf16(qa.v, kf0.v, zf);   // kv cols 0..15
        v8f s1 = wmma_bf16(qa.v, kf1.v, zf);   // kv cols 16..31

        // next block's K loads -> latency hidden under softmax + PV
        if (jb + 1 < NB) {
            load_k(kv0 + 32, kf0);
            load_k(kv0 + 48, kf1);
        }

        // ---- online softmax (per-row stats shared across 16-lane half) ----
        float sm[8];
#pragma unroll
        for (int r = 0; r < 8; ++r) sm[r] = fmaxf(s0[r], s1[r]);
#pragma unroll
        for (int off = 1; off < 16; off <<= 1)
#pragma unroll
            for (int r = 0; r < 8; ++r)
                sm[r] = fmaxf(sm[r], __shfl_xor(sm[r], off, 32));

        float sc[8], p0v[8], p1v[8], rs[8];
#pragma unroll
        for (int r = 0; r < 8; ++r) {
            const float mn = fmaxf(m[r], sm[r]);
            sc[r]  = __expf(m[r] - mn);
            m[r]   = mn;
            p0v[r] = __expf(s0[r] - mn);
            p1v[r] = __expf(s1[r] - mn);
            rs[r]  = p0v[r] + p1v[r];
        }
#pragma unroll
        for (int off = 1; off < 16; off <<= 1)
#pragma unroll
            for (int r = 0; r < 8; ++r)
                rs[r] += __shfl_xor(rs[r], off, 32);
#pragma unroll
        for (int r = 0; r < 8; ++r) l[r] = l[r] * sc[r] + rs[r];

        // rescale running O
#pragma unroll
        for (int t = 0; t < 16; ++t)
#pragma unroll
            for (int r = 0; r < 8; ++r) o[t][r] *= sc[r];

        // ---- transpose P (D-format) -> A-format via per-wave LDS ----
#pragma unroll
        for (int r = 0; r < 8; ++r) {
            const int row = r + hi * 8;
            plds[wave][row][col]      = f2bf(p0v[r]);
            plds[wave][row][16 + col] = f2bf(p1v[r]);
        }
        asm volatile("s_wait_dscnt 0" ::: "memory");
        AFrag pa;
#pragma unroll
        for (int j = 0; j < 8; ++j) {
            const int K0 = kb + 2 * j;
            pa.u[j] = *(const unsigned*)&plds[wave][col][K0];
        }

        // ---- O += P * V ----
#if HAVE_TDM
#pragma unroll
        for (int t = 0; t < 16; ++t) {
            BFrag vf;
            const uint4* p = (const uint4*)&vlds[buf][t * 16 + col][kb];
            vf.q[0] = p[0]; vf.q[1] = p[1];
            o[t] = wmma_bf16(pa.v, vf.v, o[t]);
        }
        __syncthreads();            // all waves done with tile jb before overwrite
#else
        auto load_v = [&](int t, BFrag& f) {
            const uint4* p =
                (const uint4*)(vbase_p + (size_t)(t * 16 + col) * NTOK + kv0 + kb);
            f.q[0] = p[0]; f.q[1] = p[1];
        };
        BFrag vcur, vnxt;
        load_v(0, vcur);
#pragma unroll
        for (int t = 0; t < 16; ++t) {
            if (t < 15) load_v(t + 1, vnxt);
            o[t] = wmma_bf16(pa.v, vcur.v, o[t]);
            if (t < 15) vcur = vnxt;
        }
#endif
    }

    // ---- epilogue: out = gamma * (O / l) + x, written in (B,C,H,W) layout ----
    const float g = gamma[0];
#pragma unroll
    for (int r = 0; r < 8; ++r) {
        const int np = qbase + r + hi * 8;
        const float inv_l = 1.0f / l[r];
#pragma unroll
        for (int t = 0; t < 16; ++t) {
            const int c = t * 16 + col;
            const size_t idx = ((size_t)b * CCH + c) * NTOK + np;
            out[idx] = g * (o[t][r] * inv_l) + x[idx];
        }
    }
}

// ---------------------------------------------------------------------------
extern "C" void kernel_launch(void* const* d_in, const int* in_sizes, int n_in,
                              void* d_out, int out_size, void* d_ws, size_t ws_size,
                              hipStream_t stream) {
    (void)in_sizes; (void)n_in; (void)out_size; (void)ws_size;
    const float* x     = (const float*)d_in[0];
    const float* Wq    = (const float*)d_in[1];
    const float* bq    = (const float*)d_in[2];
    const float* Wk    = (const float*)d_in[3];
    const float* bk    = (const float*)d_in[4];
    const float* Wv    = (const float*)d_in[5];
    const float* bv    = (const float*)d_in[6];
    const float* gamma = (const float*)d_in[7];

    unsigned short* qws = (unsigned short*)d_ws;              // 1 MB
    unsigned short* kws = qws + (size_t)4 * NTOK * 32;        // 1 MB
    unsigned short* vws = kws + (size_t)4 * NTOK * 32;        // 8 MB
    unsigned short* wqb = vws + (size_t)4 * CCH * NTOK;       // 16 KB
    unsigned short* wkb = wqb + 32 * CCH;                     // 16 KB
    unsigned short* wvb = wkb + 32 * CCH;                     // 128 KB

    wconv_kernel<<<128, 256, 0, stream>>>(Wq, Wk, Wv, (unsigned*)wqb,
                                          (unsigned*)wkb, (unsigned*)wvb);
    proj_kernel<<<256, 128, 0, stream>>>(x, wqb, bq, wkb, bk, wvb, bv,
                                         qws, kws, vws);
    attn_kernel<<<256, 128, 0, stream>>>(x, gamma, qws, kws, vws, (float*)d_out);
}
